// DGCNN_60679297958328
// MI455X (gfx1250) — matrix operand order repaired and checked by reading, not compiled, not run on hardware
//
#include <hip/hip_runtime.h>
#include <math.h>

// ---------------------------------------------------------------------------
// DGCNN for MI455X (gfx1250), hybrid precision:
//  - kNN gram matrices: V_WMMA_F32_16X16X4_F32 (exact fp32 -> stable top-k)
//  - EdgeConv / final GEMMs: V_WMMA_F32_16X16X32_BF16 with bf16x3 error
//    compensation (Ah*Bh + Ah*Bl + Al*Bh), ~fp32 accuracy at 2.7x fewer
//    matrix instructions per FLOP than the fp32 WMMA path.
//  - 320KB LDS/WGP holds a full 16x4096 fp32 distance strip per block;
//    top-20 selection uses cached per-lane minima + cooperative rescan.
// All blocks are 256 threads = 8 full wave32s (EXEC all-ones around WMMA).
// ---------------------------------------------------------------------------

typedef float v2f __attribute__((ext_vector_type(2)));
typedef float v8f __attribute__((ext_vector_type(8)));
typedef __bf16 bf16;
typedef bf16 v16bf __attribute__((ext_vector_type(16)));

#define BATCH 8
#define NPTS  4096
#define KNNK  20
#define HDIM  512
#define BN_EPS 1e-5f
#define SLOPE  0.2f
#define FINF 3.4e38f

__device__ __forceinline__ v8f wmma4(v2f a, v2f b, v8f c) {
  return __builtin_amdgcn_wmma_f32_16x16x4_f32(
      false, a, false, b, (short)0, c, false, false);
}
__device__ __forceinline__ v8f wmma_bf(v16bf a, v16bf b, v8f c) {
  return __builtin_amdgcn_wmma_f32_16x16x32_bf16(
      false, a, false, b, (short)0, c, false, false);
}

// ---------------------------------------------------------------------------
// (B,3,N) -> (B,N,4) padded row-major point features (pad col = 0)
__global__ void xpose_kernel(const float* __restrict__ x, float* __restrict__ xt) {
  int i = blockIdx.x * blockDim.x + threadIdx.x;
  if (i >= BATCH * NPTS) return;
  int b = i / NPTS, n = i % NPTS;
#pragma unroll
  for (int d = 0; d < 4; ++d)
    xt[(size_t)i * 4 + d] = (d < 3) ? x[((size_t)b * 3 + d) * NPTS + n] : 0.f;
}

// ---------------------------------------------------------------------------
// per-row squared norms of feature matrix (row stride S, D valid columns)
__global__ void norm_kernel(const float* __restrict__ X, int S, int D,
                            float* __restrict__ nrm) {
  int r = blockIdx.x * blockDim.x + threadIdx.x;
  if (r >= BATCH * NPTS) return;
  float s = 0.f;
  for (int d = 0; d < D; ++d) { float v = X[(size_t)r * S + d]; s += v * v; }
  nrm[r] = s;
}

// ---------------------------------------------------------------------------
// map padded feat column -> real W column (layer 1 pads D=3 -> 4)
__device__ __forceinline__ int wcol_map(int kk, int Dp, int Dt) {
  if (kk < Dp) return (kk < Dt) ? kk : -1;
  int t = kk - Dp;
  return (t < Dt) ? (Dt + t) : -1;
}

// ---------------------------------------------------------------------------
// Pre-split W into bf16 hi/lo with K zero-padded to Kp and layer-1 column
// remap folded in. For plain GEMM weights pass Dp==Dt==rowlen/2 (identity).
__global__ void wsplit_kernel(const float* __restrict__ W, int Co, int Dt, int Dp,
                              int Kp, bf16* __restrict__ hi, bf16* __restrict__ lo) {
  int i = blockIdx.x * blockDim.x + threadIdx.x;
  if (i >= Co * Kp) return;
  int o = i / Kp, kk = i % Kp;
  float v = 0.f;
  if (kk < 2 * Dp) {
    int wc = wcol_map(kk, Dp, Dt);
    if (wc >= 0) v = W[(size_t)o * (2 * Dt) + wc];
  }
  bf16 h = (bf16)v;
  hi[i] = h;
  lo[i] = (bf16)(v - (float)h);
}

// ---------------------------------------------------------------------------
// kNN: one block = 16 query rows. fp32 WMMA gram tiles -> 16x4096 dist strip
// in LDS (256KB) -> top-20 via cached per-lane minima + cooperative rescan.
__global__ __launch_bounds__(256) void knn_kernel(
    const float* __restrict__ X, int S, int D,
    const float* __restrict__ nrm, int* __restrict__ knn) {
  extern __shared__ float lds[];
  float* dist  = lds;                  // 16*4096
  float* cnorm = dist + 16 * NPTS;     // 4096
  float* At    = cnorm + NPTS;         // 16*128 (max D)
  float* rnorm = At + 16 * 128;        // 16

  const int tid = threadIdx.x;
  const int lane = tid & 31, wv = tid >> 5;
  const int b  = blockIdx.x / (NPTS / 16);
  const int pb = (blockIdx.x % (NPTS / 16)) * 16;
  const size_t rowbase = (size_t)b * NPTS;

  for (int e = tid; e < 16 * D; e += 256) {
    int m = e / D, d = e % D;
    At[m * D + d] = X[(rowbase + pb + m) * S + d];
  }
  for (int c = tid; c < NPTS; c += 256) cnorm[c] = nrm[rowbase + c];
  if (tid < 16) rnorm[tid] = nrm[rowbase + pb + tid];
  __syncthreads();

  const int m_a = lane & 15;
  const int k0  = (lane >> 4) * 2;
  const int n_b = lane & 15;

  for (int ct = wv; ct < NPTS / 16; ct += 8) {
    const int cb = ct * 16;
    v8f acc = {};
    for (int kk = 0; kk < D; kk += 4) {
      v2f a, bb;
      a.x = At[m_a * D + kk + k0];
      a.y = At[m_a * D + kk + k0 + 1];
      const float* bp = &X[(rowbase + cb + n_b) * S + kk + k0];
      bb.x = bp[0]; bb.y = bp[1];
      acc = wmma4(a, bb, acc);
    }
#pragma unroll
    for (int v = 0; v < 8; ++v) {
      int m = v + 8 * (lane >> 4);
      int n = cb + n_b;
      dist[m * NPTS + n] = rnorm[m] + cnorm[n] - 2.0f * acc[v];
    }
  }
  __syncthreads();

  // selection: wave w owns rows 2w, 2w+1
  for (int m = wv * 2; m < wv * 2 + 2; ++m) {
    float* dr = dist + m * NPTS;
    int* op = knn + (rowbase + pb + m) * KNNK;

    // per-lane cached minimum over its 128-element strided chunk
    float lmin = FINF; int lidx = 0x7fffffff;
    for (int t = 0; t < NPTS / 32; ++t) {
      int c = lane + 32 * t;
      float d = dr[c];
      if (d < lmin) { lmin = d; lidx = c; }
    }
    for (int it = 0; it < KNNK; ++it) {
      float best = lmin; int bi = lidx;
#pragma unroll
      for (int off = 16; off > 0; off >>= 1) {
        float ob = __shfl_xor(best, off, 32);
        int   oi = __shfl_xor(bi, off, 32);
        if (ob < best || (ob == best && oi < bi)) { best = ob; bi = oi; }
      }
      if (lane == 0) op[it] = bi;
      dr[bi] = FINF;                       // bi wave-uniform; mask winner
      const int wl = bi & 31;              // lane owning the winner chunk
      // cooperatively rescan winner lane's chunk for its new minimum
      float nmin = FINF; int nidx = 0x7fffffff;
      for (int t = lane; t < NPTS / 32; t += 32) {
        int c = wl + 32 * t;
        float d = dr[c];
        if (d < nmin) { nmin = d; nidx = c; }
      }
#pragma unroll
      for (int off = 16; off > 0; off >>= 1) {
        float ob = __shfl_xor(nmin, off, 32);
        int   oi = __shfl_xor(nidx, off, 32);
        if (ob < nmin || (ob == nmin && oi < nidx)) { nmin = ob; nidx = oi; }
      }
      if (lane == wl) { lmin = nmin; lidx = nidx; }
    }
  }
}

// ---------------------------------------------------------------------------
// EdgeConv: block = 16 points x all C_out. Loop over kappa: build split-bf16
// A = [xj-xi, xi] (16 x Kfp) in LDS, waves GEMM their fixed C_out tiles with
// bf16x3 WMMA, fuse BN+LeakyReLU, running max in LDS.
__global__ __launch_bounds__(256) void edge_kernel(
    const float* __restrict__ X, int S, int Dp, int Kfp,
    const bf16* __restrict__ whi, const bf16* __restrict__ wlo,
    const float* __restrict__ gamma, const float* __restrict__ beta,
    const int* __restrict__ knn, float* __restrict__ out, int Co) {
  __shared__ bf16  AfH[16 * 256];      // Kfp <= 256
  __shared__ bf16  AfL[16 * 256];
  __shared__ float accs[16 * 256];     // Co <= 256
  __shared__ int nbr[16];

  const int tid = threadIdx.x, lane = tid & 31, wv = tid >> 5;
  const int b  = blockIdx.x / (NPTS / 16);
  const int pb = (blockIdx.x % (NPTS / 16)) * 16;
  const size_t rowbase = (size_t)b * NPTS;
  const int Kf = 2 * Dp;
  const int ntiles = Co / 16;
  const float inv = rsqrtf(1.0f + BN_EPS);

  for (int e = tid; e < 16 * Co; e += 256) accs[e] = -FINF;
  __syncthreads();

  const int m_a = lane & 15;
  const int kaA = (lane >> 4) * 8;     // A half-wave K base (within 16)
  const int kbB = (lane >> 4) * 8;     // B half-wave K base (within 16)
  const int n_b = lane & 15;

  for (int kap = 0; kap < KNNK; ++kap) {
    if (tid < 16) nbr[tid] = knn[(rowbase + pb + tid) * KNNK + kap];
    __syncthreads();
    for (int e = tid; e < 16 * Kfp; e += 256) {
      int m = e / Kfp, c = e % Kfp;
      float v = 0.f;
      if (c < Kf) {
        const float* xp = &X[(rowbase + pb + m) * S];
        const float* xj = &X[(rowbase + nbr[m]) * S];
        v = (c < Dp) ? (xj[c] - xp[c]) : xp[c - Dp];
      }
      bf16 h = (bf16)v;
      AfH[e] = h;
      AfL[e] = (bf16)(v - (float)h);
    }
    __syncthreads();

    for (int nt = wv; nt < ntiles; nt += 8) {   // fixed tile->wave: no races
      const int ob = nt * 16;
      const int o = ob + n_b;
      v8f acc = {};
      for (int kk = 0; kk < Kfp; kk += 32) {
        v16bf ah, al, bh, bl;
        const bf16* arh = &AfH[m_a * Kfp + kk];
        const bf16* arl = &AfL[m_a * Kfp + kk];
#pragma unroll
        for (int j = 0; j < 8; ++j) {
          ah[j]     = arh[kaA + j];      al[j]     = arl[kaA + j];
          ah[j + 8] = arh[16 + kaA + j]; al[j + 8] = arl[16 + kaA + j];
        }
        const bf16* brh = &whi[(size_t)o * Kfp + kk + kbB];
        const bf16* brl = &wlo[(size_t)o * Kfp + kk + kbB];
#pragma unroll
        for (int j = 0; j < 8; ++j) { bh[j] = brh[j]; bl[j] = brl[j]; bh[j + 8] = brh[16 + j]; bl[j + 8] = brl[16 + j]; }
        acc = wmma_bf(ah, bh, acc);
        acc = wmma_bf(ah, bl, acc);
        acc = wmma_bf(al, bh, acc);
      }
#pragma unroll
      for (int v = 0; v < 8; ++v) {
        int m = v + 8 * (lane >> 4);
        float y = acc[v] * (gamma[o] * inv) + beta[o];
        y = (y >= 0.f) ? y : SLOPE * y;
        float* ap = &accs[m * Co + o];
        *ap = fmaxf(*ap, y);
      }
    }
    __syncthreads();
  }
  for (int e = tid; e < 16 * Co; e += 256) {
    int m = e / Co, o = e % Co;
    out[(rowbase + pb + m) * (size_t)HDIM + o] = accs[m * Co + o];
  }
}

// ---------------------------------------------------------------------------
// layer 5: y = BN_LReLU(h(32768x512) @ W5^T), bf16x3 WMMA
__global__ __launch_bounds__(256) void gemm5_kernel(
    const float* __restrict__ h, const bf16* __restrict__ whi,
    const bf16* __restrict__ wlo, const float* __restrict__ gamma,
    const float* __restrict__ beta, float* __restrict__ y) {
  __shared__ bf16 AhH[16 * HDIM];
  __shared__ bf16 AhL[16 * HDIM];
  const int tid = threadIdx.x, lane = tid & 31, wv = tid >> 5;
  const size_t rb = (size_t)blockIdx.x * 16;
  for (int e = tid; e < 16 * HDIM; e += 256) {
    float v = h[(rb + e / HDIM) * HDIM + (e % HDIM)];
    bf16 hh = (bf16)v;
    AhH[e] = hh;
    AhL[e] = (bf16)(v - (float)hh);
  }
  __syncthreads();
  const float inv = rsqrtf(1.0f + BN_EPS);
  const int m_a = lane & 15;
  const int kaA = (lane >> 4) * 8;
  const int kbB = (lane >> 4) * 8;
  const int n_b = lane & 15;
  for (int nt = wv; nt < HDIM / 16; nt += 8) {
    const int ob = nt * 16;
    const int o = ob + n_b;
    v8f acc = {};
    for (int kk = 0; kk < HDIM; kk += 32) {
      v16bf ah, al, bh, bl;
      const bf16* arh = &AhH[m_a * HDIM + kk];
      const bf16* arl = &AhL[m_a * HDIM + kk];
#pragma unroll
      for (int j = 0; j < 8; ++j) {
        ah[j]     = arh[kaA + j];      al[j]     = arl[kaA + j];
        ah[j + 8] = arh[16 + kaA + j]; al[j + 8] = arl[16 + kaA + j];
      }
      const bf16* brh = &whi[(size_t)o * HDIM + kk + kbB];
      const bf16* brl = &wlo[(size_t)o * HDIM + kk + kbB];
#pragma unroll
      for (int j = 0; j < 8; ++j) { bh[j] = brh[j]; bl[j] = brl[j]; bh[j + 8] = brh[16 + j]; bl[j + 8] = brl[16 + j]; }
      acc = wmma_bf(ah, bh, acc);
      acc = wmma_bf(ah, bl, acc);
      acc = wmma_bf(al, bh, acc);
    }
#pragma unroll
    for (int v = 0; v < 8; ++v) {
      int m = v + 8 * (lane >> 4);
      float yy = acc[v] * (gamma[o] * inv) + beta[o];
      yy = (yy >= 0.f) ? yy : SLOPE * yy;
      y[(rb + m) * HDIM + o] = yy;
    }
  }
}

// ---------------------------------------------------------------------------
// global max+mean over N per (b, o): out[b, :512]=max, out[b, 512:]=mean
__global__ __launch_bounds__(256) void reduce_kernel(
    const float* __restrict__ y, float* __restrict__ out) {
  __shared__ float smx[256], ssm[256];
  const int o = blockIdx.x % HDIM;
  const int b = blockIdx.x / HDIM;
  const int tid = threadIdx.x;
  float mx = -FINF, sm = 0.f;
  for (int n = tid; n < NPTS; n += 256) {
    float v = y[((size_t)b * NPTS + n) * HDIM + o];
    mx = fmaxf(mx, v); sm += v;
  }
  smx[tid] = mx; ssm[tid] = sm; __syncthreads();
  for (int s = 128; s > 0; s >>= 1) {
    if (tid < s) { smx[tid] = fmaxf(smx[tid], smx[tid + s]); ssm[tid] += ssm[tid + s]; }
    __syncthreads();
  }
  if (tid == 0) {
    out[(size_t)b * 2 * HDIM + o] = smx[0];
    out[(size_t)b * 2 * HDIM + HDIM + o] = ssm[0] / (float)NPTS;
  }
}

// ---------------------------------------------------------------------------
extern "C" void kernel_launch(void* const* d_in, const int* in_sizes, int n_in,
                              void* d_out, int out_size, void* d_ws, size_t ws_size,
                              hipStream_t stream) {
  (void)in_sizes; (void)n_in; (void)out_size; (void)ws_size;
  // dict order: x, k, then (W,g,b) x 5
  const float* x  = (const float*)d_in[0];
  const float* W1 = (const float*)d_in[2],  *g1 = (const float*)d_in[3],  *b1 = (const float*)d_in[4];
  const float* W2 = (const float*)d_in[5],  *g2 = (const float*)d_in[6],  *b2 = (const float*)d_in[7];
  const float* W3 = (const float*)d_in[8],  *g3 = (const float*)d_in[9],  *b3 = (const float*)d_in[10];
  const float* W4 = (const float*)d_in[11], *g4 = (const float*)d_in[12], *b4 = (const float*)d_in[13];
  const float* W5 = (const float*)d_in[14], *g5 = (const float*)d_in[15], *b5 = (const float*)d_in[16];
  float* out = (float*)d_out;

  // workspace layout (floats first, then bf16 weight splits): ~139 MB
  float* ws  = (float*)d_ws;
  float* xt  = ws;                              // B*N*4        = 131072
  float* nrm = ws + 131072;                     // B*N          = 32768
  int*   knn = (int*)(ws + 163840);             // B*N*20       = 655360
  float* h   = ws + 819200;                     // B*N*512 (x1|x2|x3|x4 @ col 0/64/128/256)
  float* y5  = h + (size_t)16777216;            // B*N*512
  bf16*  wb  = (bf16*)(y5 + (size_t)16777216);  // bf16 weight splits
  bf16 *w1h = wb,            *w1l = w1h + 64 * 32;     // Kp=32
  bf16 *w2h = w1l + 64 * 32, *w2l = w2h + 64 * 128;    // Kp=128
  bf16 *w3h = w2l + 64 * 128,*w3l = w3h + 64 * 128;
  bf16 *w4h = w3l + 64 * 128,*w4l = w4h + 256 * 256;   // Kp=256
  bf16 *w5h = w4l + 256 * 256,*w5l = w5h + 512 * 512;  // Kp=512

  const dim3 blk(256);
  const int nblk = BATCH * (NPTS / 16);
  const size_t knnLds = (size_t)(16 * NPTS + NPTS + 16 * 128 + 16) * sizeof(float);
  const int nThr = (BATCH * NPTS + 255) / 256;

  xpose_kernel<<<nThr, blk, 0, stream>>>(x, xt);

  // pre-split weights to bf16 hi/lo (layer-1 K-padding 6->32 folded in;
  // W5 uses identity map via Dp==Dt==256)
  wsplit_kernel<<<(64 * 32 + 255) / 256, blk, 0, stream>>>(W1, 64, 3, 4, 32, w1h, w1l);
  wsplit_kernel<<<(64 * 128 + 255) / 256, blk, 0, stream>>>(W2, 64, 64, 64, 128, w2h, w2l);
  wsplit_kernel<<<(64 * 128 + 255) / 256, blk, 0, stream>>>(W3, 64, 64, 64, 128, w3h, w3l);
  wsplit_kernel<<<(256 * 256 + 255) / 256, blk, 0, stream>>>(W4, 256, 128, 128, 256, w4h, w4l);
  wsplit_kernel<<<(512 * 512 + 255) / 256, blk, 0, stream>>>(W5, 512, 256, 256, 512, w5h, w5l);

  // layer 1: D=3 (padded 4) -> 64, h cols [0,64)
  norm_kernel<<<nThr, blk, 0, stream>>>(xt, 4, 4, nrm);
  knn_kernel<<<nblk, blk, knnLds, stream>>>(xt, 4, 4, nrm, knn);
  edge_kernel<<<nblk, blk, 0, stream>>>(xt, 4, 4, 32, w1h, w1l, g1, b1, knn, h + 0, 64);

  // layer 2: x1 -> 64, h cols [64,128)
  norm_kernel<<<nThr, blk, 0, stream>>>(h + 0, HDIM, 64, nrm);
  knn_kernel<<<nblk, blk, knnLds, stream>>>(h + 0, HDIM, 64, nrm, knn);
  edge_kernel<<<nblk, blk, 0, stream>>>(h + 0, HDIM, 64, 128, w2h, w2l, g2, b2, knn, h + 64, 64);

  // layer 3: x2 -> 128, h cols [128,256)
  norm_kernel<<<nThr, blk, 0, stream>>>(h + 64, HDIM, 64, nrm);
  knn_kernel<<<nblk, blk, knnLds, stream>>>(h + 64, HDIM, 64, nrm, knn);
  edge_kernel<<<nblk, blk, 0, stream>>>(h + 64, HDIM, 64, 128, w3h, w3l, g3, b3, knn, h + 128, 128);

  // layer 4: x3 -> 256, h cols [256,512)
  norm_kernel<<<nThr, blk, 0, stream>>>(h + 128, HDIM, 128, nrm);
  knn_kernel<<<nblk, blk, knnLds, stream>>>(h + 128, HDIM, 128, nrm, knn);
  edge_kernel<<<nblk, blk, 0, stream>>>(h + 128, HDIM, 128, 256, w4h, w4l, g4, b4, knn, h + 256, 256);

  // layer 5 + pooling
  gemm5_kernel<<<BATCH * NPTS / 16, blk, 0, stream>>>(h, w5h, w5l, g5, b5, y5);
  reduce_kernel<<<BATCH * HDIM, blk, 0, stream>>>(y5, out);
}
